// SteerablePyramid_46342697124237
// MI455X (gfx1250) — matrix operand order repaired and checked by reading, not compiled, not run on hardware
//
#include <hip/hip_runtime.h>
#include <hip/hip_bf16.h>

// ---------------------------------------------------------------------------
// Complex steerable pyramid on MI455X (gfx1250, wave32).
//
// FFT strategy: N = 16*M, both large stages on the matrix pipe.
//   Stage 1: DFT-16 over n1 via v_wmma_f32_16x16x32_f16, complex packed in K:
//       Dr = [Wr | -Wi] x [Vr ; Vi],  Di = [Wi | Wr] x [Vr ; Vi]
//   Stage 2 (M=32, N=512): twiddle fused into B-fragment build (incremental
//       rotation), WMMA DFT-16 over even/odd n2, then a radix-2 combine.
//   Stage 2 (M=16, N=256): twiddled WMMA DFT-16 directly.
//   Stage 2 (M<=8): scalar fused twiddle*DFT-M (small levels only).
// Memory floor ~200MB => ~10us @ 23.3 TB/s; butterflies stay on WMMA so the
// VALU is off the critical path.  Contiguous tiles ride the CDNA5 async
// LDS<->global path: global_load_async_to_lds_b128 on input,
// global_store_async_from_lds_b128 on the big inverse-FFT output.
// ---------------------------------------------------------------------------

typedef __attribute__((ext_vector_type(16))) _Float16 v16h;
typedef __attribute__((ext_vector_type(8)))  float    v8f;

#define TWOPI 6.28318530717958647692f

#define WMMA_F16(A, B, C) \
    __builtin_amdgcn_wmma_f32_16x16x32_f16(false, (A), false, (B), (short)0, (C), false, false)

// Build the A-matrix fragments for the complex DFT-16 (per-lane, wave32).
// 16-bit A 16x32 layout (contiguous, matches B): lanes 0-15 -> M=lane, element
// h holds K=h; lanes 16-31 -> same M, element h holds K=16+h.
__device__ __forceinline__ void build_dft16(float sgn, v16h& Ar, v16h& Ai) {
    const int lane = threadIdx.x & 31;
    const int m = lane & 15;
    const bool hi = lane >= 16;
#pragma unroll
    for (int h = 0; h < 16; ++h) {
        const int p = h >> 1, o = h & 1;
        int K;
        if (!hi) K = h;
        else     K = 16 + h;
        const int  j        = K & 15;      // DFT column index
        const bool imagHalf = K >= 16;     // second K-half multiplies Vi
        const float ang = sgn * TWOPI * (float)((m * j) & 15) / 16.0f;
        const float c = __cosf(ang), s = __sinf(ang);
        Ar[h] = (_Float16)(imagHalf ? -s : c);   // A_r = [Wr | -Wi]
        Ai[h] = (_Float16)(imagHalf ?  c : s);   // A_i = [Wi |  Wr]
    }
}

// Stage 1: DFT-16 over n1 (n = n1*M + n2), 16 vectors per WMMA pair.
// B 32x16 layout (wave32): lanes 0-15 hold K=0..15 (real half), lanes 16-31
// hold K=16..31 (imag half), N = lane&15.
__device__ __forceinline__ void dft16_stage1(
    const float* __restrict__ Zr, const float* __restrict__ Zi,
    float* __restrict__ Yr, float* __restrict__ Yi,
    int N, int M, const v16h& Ar, const v16h& Ai) {
    const int lane = threadIdx.x & 31;
    const int wave = threadIdx.x >> 5;
    const int n    = lane & 15;
    const bool hiHalf = lane >= 16;
    for (int b = wave; b < M; b += 8) {
        const int q   = b * 16 + n;
        const int row = q / M;
        const int g   = q - row * M;
        const float* src  = hiHalf ? Zi : Zr;
        const float* colp = src + row * N + g;        // element j at colp[j*M]
        v16h Bf;
#pragma unroll
        for (int h = 0; h < 16; ++h) Bf[h] = (_Float16)colp[h * M];
        v8f dr = {}, di = {};
        dr = WMMA_F16(Ar, Bf, dr);
        di = WMMA_F16(Ai, Bf, di);
        float* yr = Yr + row * N + g;
        float* yi = Yi + row * N + g;
        const int mbase = hiHalf ? 8 : 0;
#pragma unroll
        for (int v = 0; v < 8; ++v) {
            yr[(mbase + v) * M] = dr[v];
            yi[(mbase + v) * M] = di[v];
        }
    }
}

// Stage 2 for N=256 (M=16): X[k1+16k2] = DFT16_{n2}( Y[k1][n2]*W_256^{k1 n2} ).
__device__ __forceinline__ void dft16_stage2_m16(
    const float* __restrict__ Yr, const float* __restrict__ Yi,
    float* __restrict__ Zr, float* __restrict__ Zi,
    float sgn, const v16h& Ar, const v16h& Ai) {
    const int N = 256;
    const int lane = threadIdx.x & 31;
    const int wave = threadIdx.x >> 5;
    const int n    = lane & 15;
    const bool hiHalf = lane >= 16;
    for (int b = wave; b < 16; b += 8) {
        const int v   = b * 16 + n;
        const int row = v >> 4;
        const int k1  = v & 15;
        const float* yr = Yr + row * N + k1 * 16;
        const float* yi = Yi + row * N + k1 * 16;
        float sd, cd;
        __sincosf(sgn * TWOPI * (float)k1 / 256.f, &sd, &cd);
        float cr = 1.f, ci = 0.f;
        v16h Bf;
#pragma unroll
        for (int h = 0; h < 16; ++h) {
            const float ar = yr[h], ai = yi[h];
            Bf[h] = (_Float16)(hiHalf ? fmaf(ar, ci,  ai * cr)
                                      : fmaf(ar, cr, -ai * ci));
            const float nc = cr * cd - ci * sd;
            ci = cr * sd + ci * cd;
            cr = nc;
        }
        v8f dr = {}, di = {};
        dr = WMMA_F16(Ar, Bf, dr);
        di = WMMA_F16(Ai, Bf, di);
        float* zr = Zr + row * N + k1;     // X at [row][16*k2 + k1]
        float* zi = Zi + row * N + k1;
        const int mbase = hiHalf ? 8 : 0;
#pragma unroll
        for (int vv = 0; vv < 8; ++vv) {
            zr[(mbase + vv) * 16] = dr[vv];
            zi[(mbase + vv) * 16] = di[vv];
        }
    }
}

// Stage 2a for N=512 (M=32): per (row,k1,parity p) twiddled DFT-16 over
// n2 = 2j+p.  Results E/O stored at [row][k1*32 + p*16 + m].
__device__ __forceinline__ void dft16_stage2_m32(
    const float* __restrict__ Yr, const float* __restrict__ Yi,
    float* __restrict__ Zr, float* __restrict__ Zi,
    float sgn, const v16h& Ar, const v16h& Ai) {
    const int lane = threadIdx.x & 31;
    const int wave = threadIdx.x >> 5;
    const int n    = lane & 15;
    const bool hiHalf = lane >= 16;
    for (int b = wave; b < 32; b += 8) {
        const int v   = b * 16 + n;
        const int row = v >> 5;
        const int rem = v & 31;
        const int k1  = rem >> 1;
        const int p   = rem & 1;
        const float* yr = Yr + row * 512 + k1 * 32 + p;
        const float* yi = Yi + row * 512 + k1 * 32 + p;
        float sd, cd;
        __sincosf(sgn * TWOPI * (float)k1 / 256.f, &sd, &cd);   // step: n2 += 2
        float cr, ci;
        __sincosf(sgn * TWOPI * (float)(k1 * p) / 512.f, &ci, &cr);  // phase n2=p
        v16h Bf;
#pragma unroll
        for (int h = 0; h < 16; ++h) {
            const float ar = yr[2 * h], ai = yi[2 * h];
            Bf[h] = (_Float16)(hiHalf ? fmaf(ar, ci,  ai * cr)
                                      : fmaf(ar, cr, -ai * ci));
            const float nc = cr * cd - ci * sd;
            ci = cr * sd + ci * cd;
            cr = nc;
        }
        v8f dr = {}, di = {};
        dr = WMMA_F16(Ar, Bf, dr);
        di = WMMA_F16(Ai, Bf, di);
        float* zr = Zr + row * 512 + k1 * 32 + p * 16;
        float* zi = Zi + row * 512 + k1 * 32 + p * 16;
        const int mbase = hiHalf ? 8 : 0;
#pragma unroll
        for (int vv = 0; vv < 8; ++vv) {
            zr[mbase + vv] = dr[vv];
            zi[mbase + vv] = di[vv];
        }
    }
}

// Stage 2b for N=512: radix-2 combine, X[k1+16k2] = E[m] +/- W32^{k2} O[m].
__device__ __forceinline__ void radix2_combine_512(
    const float* __restrict__ Er, const float* __restrict__ Ei,
    float* __restrict__ Xr, float* __restrict__ Xi, float sgn) {
    for (int idx = threadIdx.x; idx < 4096; idx += blockDim.x) {
        const int row = idx >> 8;
        const int k1  = (idx >> 4) & 15;
        const int m   = idx & 15;
        const int be  = row * 512 + k1 * 32 + m;
        const float er = Er[be],      ei = Ei[be];
        const float pr = Er[be + 16], pi = Ei[be + 16];
        float s, c;
        __sincosf(sgn * TWOPI * (float)m / 32.f, &s, &c);
        const float tr = pr * c - pi * s;
        const float ti = pr * s + pi * c;
        const int klo = row * 512 + k1 + 16 * m;
        Xr[klo] = er + tr;        Xi[klo] = ei + ti;
        Xr[klo + 256] = er - tr;  Xi[klo + 256] = ei - ti;
    }
}

// Full N-point complex FFT (N = 16*M) on a 16-row tile in LDS.
// Returns 1 if the result ends in Yr/Yi, 0 if in Zr/Zi.
__device__ __forceinline__ int fft_tile(float* Zr, float* Zi,
                                        float* Yr, float* Yi,
                                        int N, float sgn,
                                        const v16h& Ar, const v16h& Ai) {
    const int M = N >> 4;
    __syncthreads();
    dft16_stage1(Zr, Zi, Yr, Yi, N, M, Ar, Ai);
    __syncthreads();
    if (M == 32) {
        dft16_stage2_m32(Yr, Yi, Zr, Zi, sgn, Ar, Ai);
        __syncthreads();
        radix2_combine_512(Zr, Zi, Yr, Yi, sgn);
        __syncthreads();
        return 1;
    }
    if (M == 16) {
        dft16_stage2_m16(Yr, Yi, Zr, Zi, sgn, Ar, Ai);
        __syncthreads();
        return 0;
    }
    // scalar stage 2 for small levels (M <= 8)
    for (int idx = threadIdx.x; idx < 16 * N; idx += blockDim.x) {
        const int row = idx / N;
        const int k   = idx - row * N;
        const int k1  = k & 15;
        float sd, cd;
        __sincosf(sgn * TWOPI * (float)k / (float)N, &sd, &cd);
        const float* yr = Yr + row * N + k1 * M;
        const float* yi = Yi + row * N + k1 * M;
        float cr = 1.f, ci = 0.f, xr = 0.f, xi = 0.f;
        for (int n2 = 0; n2 < M; ++n2) {
            const float ar = yr[n2], ai = yi[n2];
            xr = fmaf(ar, cr, fmaf(-ai, ci, xr));
            xi = fmaf(ar, ci, fmaf( ai, cr, xi));
            const float nc = cr * cd - ci * sd;
            ci = cr * sd + ci * cd;
            cr = nc;
        }
        Zr[row * N + k] = xr;
        Zi[row * N + k] = xi;
    }
    __syncthreads();
    return 0;
}

// --------------------------- forward kernels -------------------------------

// Row FFT of real input: one block = 16 rows of one (b,c) image.
__global__ void k_row_fft_fwd(const float* __restrict__ x,
                              float2* __restrict__ S0) {
    extern __shared__ float smem[];
    const int N = 512, WP = 272;
    float* Zr = smem;                 // NOTE: at dynamic-LDS offset 0
    float* Zi = Zr + 16 * N;
    float* Yr = Zi + 16 * N;
    float* Yi = Yr + 16 * N;
    const int tile = blockIdx.x, bc = blockIdx.y;
    const float* src = x + ((long)bc * 512 + (long)tile * 16) * 512;
#if defined(__HIP_DEVICE_COMPILE__)
    // Async-copy the contiguous 32KB tile straight into LDS (ASYNCcnt path).
    for (int i = threadIdx.x; i < (16 * 512) / 4; i += blockDim.x) {
        const unsigned ldsOff = (unsigned)(i * 16u);   // Zr region base == 0
        const float* ga = src + i * 4;
        asm volatile("global_load_async_to_lds_b128 %0, %1, off"
                     :: "v"(ldsOff), "v"(ga) : "memory");
    }
    asm volatile("s_wait_asynccnt 0" ::: "memory");
#else
    for (int i = threadIdx.x; i < 16 * N; i += blockDim.x) Zr[i] = src[i];
#endif
    for (int i = threadIdx.x; i < 16 * N; i += blockDim.x) Zi[i] = 0.f;
    __syncthreads();
    v16h Ar, Ai;
    build_dft16(-1.f, Ar, Ai);
    const int flip = fft_tile(Zr, Zi, Yr, Yi, N, -1.f, Ar, Ai);
    const float* Rr = flip ? Yr : Zr;
    const float* Ri = flip ? Yi : Zi;
    float2* dst = S0 + ((long)bc * 512 + (long)tile * 16) * WP;
    for (int i = threadIdx.x; i < 16 * WP; i += blockDim.x) {
        const int r = i / WP, k = i - r * WP;
        dst[i] = make_float2(Rr[r * N + k], Ri[r * N + k]);
    }
}

// Column FFT (512-pt complex) over 16-column tiles; stores fftshifted rows.
__global__ void k_col_fft_fwd(const float2* __restrict__ S0,
                              float2* __restrict__ S1) {
    extern __shared__ float smem[];
    const int N = 512, WP = 272;
    float* Zr = smem;
    float* Zi = Zr + 16 * N;
    float* Yr = Zi + 16 * N;
    float* Yi = Yr + 16 * N;
    const int ct = blockIdx.x, bc = blockIdx.y;
    const int col0 = ct * 16;
    const float2* src = S0 + (long)bc * 512 * WP;
    for (int i = threadIdx.x; i < 16 * N; i += blockDim.x) {
        const int c = i / N, r = i - c * N;
        const float2 v = src[(long)r * WP + col0 + c];
        Zr[c * N + r] = v.x;
        Zi[c * N + r] = v.y;
    }
    v16h Ar, Ai;
    build_dft16(-1.f, Ar, Ai);
    const int flip = fft_tile(Zr, Zi, Yr, Yi, N, -1.f, Ar, Ai);
    const float* Rr = flip ? Yr : Zr;
    const float* Ri = flip ? Yi : Zi;
    float2* dst = S1 + (long)bc * 512 * WP;
    for (int i = threadIdx.x; i < 16 * N; i += blockDim.x) {
        const int c = i / N, k = i - c * N;
        const int ks = (k + 256) & 511;   // fftshift along rows
        dst[(long)ks * WP + col0 + c] = make_float2(Rr[c * N + k], Ri[c * N + k]);
    }
}

// --------------------------- elementwise kernels ---------------------------

__global__ void k_mul_band(float2* __restrict__ dst,
                           const float2* __restrict__ src,
                           const float* __restrict__ filt,
                           int H, int wf, int wp) {
    const long plane = (long)H * wp;
    const long total = 24L * plane;
    for (long i = (long)blockIdx.x * blockDim.x + threadIdx.x; i < total;
         i += (long)gridDim.x * blockDim.x) {
        const long rem = i % plane;
        const int r = (int)(rem / wp);
        const int j = (int)(rem - (long)r * wp);
        float2 v = make_float2(0.f, 0.f);
        if (j < wf) {
            __builtin_prefetch(filt + (long)r * wf + j, 0, 0);
            const float f = filt[(long)r * wf + j];
            const float2 s = src[i];
            v = make_float2(f * s.x, f * s.y);
        }
        dst[i] = v;
    }
}

__global__ void k_lf_next(float2* __restrict__ dst,
                          const float2* __restrict__ src,
                          const float* __restrict__ l,
                          int Hs, int wpS, int wpD) {
    const int Hd  = Hs >> 1;
    const int wfD = (Hs >> 2) + 1;
    const long planeD = (long)Hd * wpD;
    const long total  = 24L * planeD;
    for (long i = (long)blockIdx.x * blockDim.x + threadIdx.x; i < total;
         i += (long)gridDim.x * blockDim.x) {
        const long bc  = i / planeD;
        const long rem = i - bc * planeD;
        const int r = (int)(rem / wpD);
        const int j = (int)(rem - (long)r * wpD);
        float2 v = make_float2(0.f, 0.f);
        if (j < wfD) {
            const float f = l[(long)r * wfD + j];
            const float2 s =
                src[bc * (long)Hs * wpS + (long)(r + (Hs >> 2)) * wpS + j];
            v = make_float2(f * s.x, f * s.y);
        }
        dst[i] = v;
    }
}

// --------------------------- inverse kernels -------------------------------

__global__ void k_col_ifft(const float2* __restrict__ T1,
                           float2* __restrict__ T2,
                           int H, int wf, int wp) {
    extern __shared__ float smem[];
    const int N = H;
    float* Zr = smem;
    float* Zi = Zr + 16 * N;
    float* Yr = Zi + 16 * N;
    float* Yi = Yr + 16 * N;
    const int ct = blockIdx.x, bc = blockIdx.y;
    const int col0 = ct * 16;
    const float2* src = T1 + (long)bc * H * wp;
    for (int i = threadIdx.x; i < 16 * N; i += blockDim.x) {
        const int c = i / N, r = i - c * N;
        const int col = col0 + c;
        float2 v = make_float2(0.f, 0.f);
        if (col < wf) v = src[(long)((r + (H >> 1)) & (H - 1)) * wp + col];
        Zr[c * N + r] = v.x;
        Zi[c * N + r] = v.y;
    }
    v16h Ar, Ai;
    build_dft16(1.f, Ar, Ai);
    const int flip = fft_tile(Zr, Zi, Yr, Yi, N, 1.f, Ar, Ai);
    const float* Rr = flip ? Yr : Zr;
    const float* Ri = flip ? Yi : Zi;
    const float invH = 1.f / (float)H;
    float2* dst = T2 + (long)bc * H * wp;
    for (int i = threadIdx.x; i < 16 * N; i += blockDim.x) {
        const int c = i / N, k = i - c * N;
        const int col = col0 + c;
        if (col < wf)
            dst[(long)k * wp + col] =
                make_float2(Rr[c * N + k] * invH, Ri[c * N + k] * invH);
    }
}

__global__ void k_row_irfft(const float2* __restrict__ T2,
                            float* __restrict__ outBase,
                            int H, int wf, int wp, long bcStride) {
    extern __shared__ float smem[];
    const int N = H;
    float* Zr = smem;
    float* Zi = Zr + 16 * N;
    float* Yr = Zi + 16 * N;
    float* Yi = Yr + 16 * N;
    const int tile = blockIdx.x, bc = blockIdx.y;
    const float2* src = T2 + (long)bc * H * wp + (long)tile * 16 * wp;
    for (int i = threadIdx.x; i < 16 * N; i += blockDim.x) {
        const int r = i / N, j = i - r * N;
        float2 v;
        if (j < wf) {
            v = src[(long)r * wp + j];
        } else {
            const float2 m = src[(long)r * wp + (N - j)];
            v = make_float2(m.x, -m.y);
        }
        Zr[r * N + j] = v.x;
        Zi[r * N + j] = v.y;
    }
    v16h Ar, Ai;
    build_dft16(1.f, Ar, Ai);
    const int flip = fft_tile(Zr, Zi, Yr, Yi, N, 1.f, Ar, Ai);
    float* Rw = flip ? Yr : Zr;
    const float invH = 1.f / (float)H;
    // fold 1/H into the LDS-resident result, then async-store the contiguous
    // 16xH real tile straight from LDS to global (STOREcnt/ASYNCcnt path).
    for (int i = threadIdx.x; i < 16 * N; i += blockDim.x) Rw[i] *= invH;
    __syncthreads();
    float* dst = outBase + bcStride * bc + (long)tile * 16 * H;
#if defined(__HIP_DEVICE_COMPILE__)
    const unsigned base = (unsigned)((Rw - smem) * sizeof(float));
    for (int i = threadIdx.x; i < (16 * N) / 4; i += blockDim.x) {
        const unsigned ldsOff = base + (unsigned)(i * 16);
        float* ga = dst + i * 4;
        asm volatile("global_store_async_from_lds_b128 %0, %1, off"
                     :: "v"(ga), "v"(ldsOff) : "memory");
    }
    asm volatile("s_wait_asynccnt 0" ::: "memory");
#else
    for (int i = threadIdx.x; i < 16 * N; i += blockDim.x) dst[i] = Rw[i];
#endif
}

// ------------------------------- host side ---------------------------------

static inline int wpOf(int H) {
    const int wf = H / 2 + 1;
    return (wf + 15) / 16 * 16;
}
static inline size_t shBytes(int H) { return (size_t)4 * 16 * H * sizeof(float); }
static inline int ewBlocks(long total) {
    long b = (total + 255) / 256;
    if (b > 32768) b = 32768;
    return (int)b;
}

extern "C" void kernel_launch(void* const* d_in, const int* in_sizes, int n_in,
                              void* d_out, int out_size, void* d_ws, size_t ws_size,
                              hipStream_t stream) {
    (void)in_sizes; (void)n_in; (void)out_size; (void)ws_size;
    const float* x   = (const float*)d_in[0];
    const float* hl0 = (const float*)d_in[1];
    const float* bF[4] = { (const float*)d_in[2], (const float*)d_in[4],
                           (const float*)d_in[6], (const float*)d_in[8] };
    const float* lF[4] = { (const float*)d_in[3], (const float*)d_in[5],
                           (const float*)d_in[7], (const float*)d_in[9] };
    float* out = (float*)d_out;

    const int BC = 24;
    const long szBig = (long)BC * 512 * 272;
    float2* S0 = (float2*)d_ws;
    float2* S1 = S0 + szBig;
    float2* T1 = S1 + szBig;
    float2* T2 = T1 + szBig;
    float2* LF[5];
    LF[0] = T2 + szBig;
    LF[1] = LF[0] + (long)BC * 512 * 272;
    LF[2] = LF[1] + (long)BC * 256 * 144;
    LF[3] = LF[2] + (long)BC * 128 * 80;
    LF[4] = LF[3] + (long)BC * 64 * 48;

    const long o0 = 0;
    const long o1 = o0 + 24L * 512 * 512;
    const long o2 = o1 + 24L * 4 * 512 * 512;
    const long o3 = o2 + 24L * 4 * 256 * 256;
    const long o4 = o3 + 24L * 4 * 128 * 128;
    const long o5 = o4 + 24L * 4 * 64 * 64;
    const long oLevel[4] = { o1, o2, o3, o4 };

    // ---- forward 2D rFFT (shifted spectrum in S1) ----
    k_row_fft_fwd<<<dim3(32, BC), 256, shBytes(512), stream>>>(x, S0);
    k_col_fft_fwd<<<dim3(17, BC), 256, shBytes(512), stream>>>(S0, S1);

    // ---- lowpass chain seed: LF0 = hl0[1] * X ----
    k_mul_band<<<ewBlocks(24L * 512 * 272), 256, 0, stream>>>(
        LF[0], S1, hl0 + 512L * 257, 512, 257, 272);

    // ---- h0 band ----
    k_mul_band<<<ewBlocks(24L * 512 * 272), 256, 0, stream>>>(
        T1, S1, hl0, 512, 257, 272);
    k_col_ifft<<<dim3(17, BC), 256, shBytes(512), stream>>>(T1, T2, 512, 257, 272);
    k_row_irfft<<<dim3(32, BC), 256, shBytes(512), stream>>>(
        T2, out + o0, 512, 257, 272, 512L * 512);

    // ---- oriented bands per level + lowpass recursion ----
    for (int n = 0; n < 4; ++n) {
        const int H  = 512 >> n;
        const int wf = H / 2 + 1;
        const int wp = wpOf(H);
        const long bcStride = 4L * H * H;
        for (int band = 0; band < 4; ++band) {
            k_mul_band<<<ewBlocks(24L * H * wp), 256, 0, stream>>>(
                T1, LF[n], bF[n] + (long)band * H * wf, H, wf, wp);
            k_col_ifft<<<dim3((wf + 15) / 16, BC), 256, shBytes(H), stream>>>(
                T1, T2, H, wf, wp);
            k_row_irfft<<<dim3(H / 16, BC), 256, shBytes(H), stream>>>(
                T2, out + oLevel[n] + (long)band * H * H, H, wf, wp, bcStride);
        }
        k_lf_next<<<ewBlocks(24L * (H / 2) * wpOf(H / 2)), 256, 0, stream>>>(
            LF[n + 1], LF[n], lF[n], H, wp, wpOf(H / 2));
    }

    // ---- residual lowpass ----
    k_col_ifft<<<dim3(2, BC), 256, shBytes(32), stream>>>(LF[4], T2, 32, 17, 32);
    k_row_irfft<<<dim3(2, BC), 256, shBytes(32), stream>>>(
        T2, out + o5, 32, 17, 32, 32L * 32);
}
